// GATNet_mse_53575422051023
// MI455X (gfx1250) — compile-verified, matrix operand order checked by reading, not run on hardware
//
#include <hip/hip_runtime.h>
#include <stdint.h>

#define N_NODES 20000
#define N_EDGES 320000
#define NE_TOT  (N_EDGES + N_NODES)
#define HD      256
#define NLAYERS 6
#define NG      50
#define BN_EPS  1e-5f
#define SLOPE   0.2f

typedef __attribute__((ext_vector_type(16))) __bf16 bf16x16;
typedef __attribute__((ext_vector_type(8)))  float  f32x8;
typedef __attribute__((ext_vector_type(8)))  int    i32x8;

union FragAB { i32x8 i; bf16x16 b; };

static __device__ __forceinline__ unsigned short f2bf(float f) {
    unsigned u = __float_as_uint(f);
    unsigned r = u + 0x7FFFu + ((u >> 16) & 1u);   // round-to-nearest-even
    return (unsigned short)(r >> 16);
}
static __device__ __forceinline__ unsigned ordEnc(float f) {
    unsigned u = __float_as_uint(f);
    return (u & 0x80000000u) ? ~u : (u | 0x80000000u);
}
static __device__ __forceinline__ float ordDec(unsigned e) {
    return __uint_as_float((e & 0x80000000u) ? (e & 0x7FFFFFFFu) : ~e);
}

// ---- pack fp32 weights (256x256, row-major [k][n]) into WMMA B-fragment layout (bf16) ----
// layout: [kt(8)][nt(16)][lane(32)][elem(16)]  (matches 16-bit operand layout, lane = column)
__global__ void k_pack_w(const float* W1, const float* W2, const float* W3,
                         const float* W4, const float* W5, unsigned short* wpk) {
    int layer = blockIdx.y;
    int t = blockIdx.x * 256 + threadIdx.x;        // 0..65535
    const float* Ws[5] = {W1, W2, W3, W4, W5};
    const float* W = Ws[layer];
    int e    = t & 15;
    int lane = (t >> 4) & 31;
    int nt   = (t >> 9) & 15;
    int kt   = t >> 13;
    int kk = (lane >> 4) << 3;
    int kl = (e < 8) ? (kk + e) : (16 + kk + (e - 8));
    int k = kt * 32 + kl;
    int n = nt * 16 + (lane & 15);
    wpk[layer * 65536 + t] = f2bf(W[k * HD + n]);
}

// ---- layer 0: h = x(N x 4) @ W0(4 x 256) ----
__global__ void k_gemm0(const float* __restrict__ x, const float* __restrict__ W0,
                        float* __restrict__ h) {
    int i = blockIdx.x * 256 + threadIdx.x;
    if (i >= N_NODES * HD) return;
    int n = i >> 8, c = i & 255;
    float s = x[n * 4 + 0] * W0[0 * HD + c] + x[n * 4 + 1] * W0[1 * HD + c]
            + x[n * 4 + 2] * W0[2 * HD + c] + x[n * 4 + 3] * W0[3 * HD + c];
    h[i] = s;
}

__global__ void k_cvt_bf16(const float* __restrict__ x, unsigned short* __restrict__ xb) {
    int i = blockIdx.x * 256 + threadIdx.x;
    if (i < N_NODES * HD) xb[i] = f2bf(x[i]);
}

// ---- WMMA GEMM: C(N x 256) = A(N x 256, bf16 row-major) @ Wpk (pre-packed) ----
// one wave per block; block computes a 16x64 tile. grid = (N/16, 4)
__global__ void k_gemm_wmma(const unsigned short* __restrict__ A,
                            const unsigned short* __restrict__ Wl,
                            float* __restrict__ C) {
    int lane = threadIdx.x;
    int rowBase = blockIdx.x * 16;
    f32x8 acc[4];
    for (int j = 0; j < 4; ++j)
        for (int q = 0; q < 8; ++q) acc[j][q] = 0.0f;

    int m  = rowBase + (lane & 15);
    int kk = (lane >> 4) << 3;          // 0 or 8
    for (int kt = 0; kt < 8; ++kt) {
        FragAB fa;
        const uint4* p0 = (const uint4*)(A + m * HD + kt * 32 + kk);
        const uint4* p1 = (const uint4*)(A + m * HD + kt * 32 + 16 + kk);
        uint4 lo = *p0, hi = *p1;
        fa.i[0] = (int)lo.x; fa.i[1] = (int)lo.y; fa.i[2] = (int)lo.z; fa.i[3] = (int)lo.w;
        fa.i[4] = (int)hi.x; fa.i[5] = (int)hi.y; fa.i[6] = (int)hi.z; fa.i[7] = (int)hi.w;
        for (int j = 0; j < 4; ++j) {
            int nt = blockIdx.y * 4 + j;
            FragAB fb;
            const i32x8* bp = (const i32x8*)(Wl + (size_t)(kt * 16 + nt) * 512);
            fb.i = bp[lane];
            acc[j] = __builtin_amdgcn_wmma_f32_16x16x32_bf16(
                false, fa.b, false, fb.b, (short)0, acc[j], false, false);
        }
    }
    int rowOff = (lane >> 4) << 3;      // 0 or 8
    int nCol   = lane & 15;
    for (int j = 0; j < 4; ++j) {
        int colBase = (blockIdx.y * 4 + j) * 16 + nCol;
        for (int r = 0; r < 8; ++r)
            C[(size_t)(rowBase + rowOff + r) * HD + colBase] = acc[j][r];
    }
}

// ---- attention scores: aS[n]=h[n,:].a_src, aD[n]=h[n,:].a_dst (wave per node) ----
__global__ void k_dots(const float* __restrict__ h, const float* __restrict__ wsrc,
                       const float* __restrict__ wdst, float* __restrict__ aS,
                       float* __restrict__ aD) {
    int wid = threadIdx.x >> 5, lane = threadIdx.x & 31;
    int node = blockIdx.x * 8 + wid;
    if (node >= N_NODES) return;
    float s0 = 0.f, s1 = 0.f;
    for (int c = 0; c < 8; ++c) {
        int col = c * 32 + lane;
        float v = h[(size_t)node * HD + col];
        s0 += v * wsrc[col];
        s1 += v * wdst[col];
    }
    for (int o = 16; o > 0; o >>= 1) {
        s0 += __shfl_xor(s0, o);
        s1 += __shfl_xor(s1, o);
    }
    if (lane == 0) { aS[node] = s0; aD[node] = s1; }
}

__global__ void k_prep(float* __restrict__ agg, unsigned* __restrict__ mOrd,
                       float* __restrict__ denom) {
    int i = blockIdx.x * 256 + threadIdx.x;
    if (i < N_NODES * HD) agg[i] = 0.f;
    if (i < N_NODES) { mOrd[i] = 0u; denom[i] = 0.f; }
}

__global__ void k_edge_att(const int* __restrict__ ei, const float* __restrict__ aS,
                           const float* __restrict__ aD, float* __restrict__ att,
                           unsigned* __restrict__ mOrd) {
    int e = blockIdx.x * 256 + threadIdx.x;
    if (e >= NE_TOT) return;
    int s, d;
    if (e < N_EDGES) { s = ei[e]; d = ei[N_EDGES + e]; } else { s = d = e - N_EDGES; }
    float a = aS[s] + aD[d];
    a = (a > 0.f) ? a : SLOPE * a;
    att[e] = a;
    atomicMax(&mOrd[d], ordEnc(a));
}

__global__ void k_edge_w(const int* __restrict__ ei, float* __restrict__ att,
                         const unsigned* __restrict__ mOrd, float* __restrict__ denom) {
    int e = blockIdx.x * 256 + threadIdx.x;
    if (e >= NE_TOT) return;
    int d = (e < N_EDGES) ? ei[N_EDGES + e] : (e - N_EDGES);
    float w = __expf(att[e] - ordDec(mOrd[d]));
    att[e] = w;
    atomicAdd(&denom[d], w);
}

// ---- out[dst,:] += (w/denom[dst]) * h[src,:]  (wave per edge) ----
__global__ void k_aggr(const int* __restrict__ ei, const float* __restrict__ w,
                       const float* __restrict__ denom, const float* __restrict__ h,
                       float* __restrict__ out) {
    int e = blockIdx.x * 8 + (threadIdx.x >> 5);
    if (e >= NE_TOT) return;
    int lane = threadIdx.x & 31;
    int s, d;
    if (e < N_EDGES) { s = ei[e]; d = ei[N_EDGES + e]; } else { s = d = e - N_EDGES; }
    float alpha = w[e] / denom[d];
    for (int c = 0; c < 8; ++c) {
        int col = c * 32 + lane;
        atomicAdd(&out[(size_t)d * HD + col], alpha * h[(size_t)s * HD + col]);
    }
}

__global__ void k_bias_relu(float* __restrict__ x, const float* __restrict__ b,
                            float* __restrict__ colsum, float* __restrict__ colsq) {
    int i = blockIdx.x * 256 + threadIdx.x;
    if (i >= N_NODES * HD) return;
    float v = x[i] + b[i & 255];
    x[i] = fmaxf(v, 0.f);
    if (i < HD) { colsum[i] = 0.f; colsq[i] = 0.f; }
}

__global__ void k_bn_stats(const float* __restrict__ x, float* __restrict__ colsum,
                           float* __restrict__ colsq) {
    int t = threadIdx.x;
    float s = 0.f, q = 0.f;
    for (int r = blockIdx.x; r < N_NODES; r += gridDim.x) {
        float v = x[(size_t)r * HD + t];
        s += v; q += v * v;
    }
    atomicAdd(&colsum[t], s);
    atomicAdd(&colsq[t], q);
}

__global__ void k_bn_apply(float* __restrict__ x, const float* __restrict__ colsum,
                           const float* __restrict__ colsq, const float* __restrict__ gamma,
                           const float* __restrict__ beta) {
    int i = blockIdx.x * 256 + threadIdx.x;
    if (i >= N_NODES * HD) return;
    int c = i & 255;
    const float invN = 1.0f / (float)N_NODES;
    float mean = colsum[c] * invN;
    float var  = colsq[c] * invN - mean * mean;
    x[i] = gamma[c] * (x[i] - mean) * rsqrtf(var + BN_EPS) + beta[c];
}

__global__ void k_zero_pool(float* __restrict__ pooled, float* __restrict__ counts) {
    int i = blockIdx.x * 256 + threadIdx.x;
    if (i < NG * HD) pooled[i] = 0.f;
    if (i < NG) counts[i] = 0.f;
}

__global__ void k_pool(const float* __restrict__ x, const int* __restrict__ batch,
                       float* __restrict__ pooled, float* __restrict__ counts) {
    int t = threadIdx.x;
    for (int r = blockIdx.x; r < N_NODES; r += gridDim.x) {
        int g = batch[r];
        atomicAdd(&pooled[(size_t)g * HD + t], x[(size_t)r * HD + t]);
        if (t == 0) atomicAdd(&counts[g], 1.0f);
    }
}

__global__ void k_pool_div(float* __restrict__ pooled, const float* __restrict__ counts) {
    int i = blockIdx.x * 256 + threadIdx.x;
    if (i >= NG * HD) return;
    pooled[i] /= fmaxf(counts[i >> 8], 1.0f);
}

__global__ void k_dense(const float* __restrict__ pooled, const float* __restrict__ dW,
                        const float* __restrict__ db, float* __restrict__ hd) {
    __shared__ float sp[HD];
    int g = blockIdx.x, t = threadIdx.x;
    sp[t] = pooled[(size_t)g * HD + t];
    __syncthreads();
    float s = 0.f;
    for (int k = 0; k < HD; ++k) s += sp[k] * dW[(size_t)k * HD + t];
    hd[(size_t)g * HD + t] = fmaxf(s + db[t], 0.f);
}

__global__ void k_mu(const float* __restrict__ hd, const float* __restrict__ muW,
                     const float* __restrict__ muB, float* __restrict__ out) {
    __shared__ float red[HD];
    int g = blockIdx.x, t = threadIdx.x;
    red[t] = hd[(size_t)g * HD + t] * muW[t];
    __syncthreads();
    for (int o = 128; o > 0; o >>= 1) {
        if (t < o) red[t] += red[t + o];
        __syncthreads();
    }
    if (t == 0) out[g] = red[0] + muB[0];
}

extern "C" void kernel_launch(void* const* d_in, const int* in_sizes, int n_in,
                              void* d_out, int out_size, void* d_ws, size_t ws_size,
                              hipStream_t stream) {
    (void)in_sizes; (void)n_in; (void)out_size; (void)ws_size;

    const float* x_in  = (const float*)d_in[0];
    const int*   ei    = (const int*)d_in[1];
    const int*   batch = (const int*)d_in[2];
    // params: layer l base = 3 + 6*l : W, a_src, a_dst, b, gamma, beta
    const float* dW  = (const float*)d_in[3 + 6 * NLAYERS + 0];
    const float* db  = (const float*)d_in[3 + 6 * NLAYERS + 1];
    const float* muW = (const float*)d_in[3 + 6 * NLAYERS + 2];
    const float* muB = (const float*)d_in[3 + 6 * NLAYERS + 3];
    float* out = (float*)d_out;

    char* ws = (char*)d_ws;
    size_t off = 0;
    auto take = [&](size_t bytes) { size_t r = off; off = (off + bytes + 255) & ~(size_t)255; return r; };
    float*          bufX   = (float*)(ws + take((size_t)N_NODES * HD * 4));
    float*          bufH   = (float*)(ws + take((size_t)N_NODES * HD * 4));
    unsigned short* bufHbf = (unsigned short*)(ws + take((size_t)N_NODES * HD * 2));
    unsigned short* wpk    = (unsigned short*)(ws + take((size_t)5 * 65536 * 2));
    float*          aS     = (float*)(ws + take((size_t)N_NODES * 4));
    float*          aD     = (float*)(ws + take((size_t)N_NODES * 4));
    unsigned*       mOrd   = (unsigned*)(ws + take((size_t)N_NODES * 4));
    float*          denom  = (float*)(ws + take((size_t)N_NODES * 4));
    float*          attw   = (float*)(ws + take((size_t)NE_TOT * 4));
    float*          colsum = (float*)(ws + take((size_t)HD * 4));
    float*          colsq  = (float*)(ws + take((size_t)HD * 4));
    float*          pooled = (float*)(ws + take((size_t)NG * HD * 4));
    float*          counts = (float*)(ws + take((size_t)NG * 4));
    float*          hdbuf  = (float*)(ws + take((size_t)NG * HD * 4));

    const int NH_BLOCKS = (N_NODES * HD) / 256;             // 20000
    const int NE_BLOCKS = (NE_TOT + 255) / 256;

    // pack layer-1..5 weights into WMMA B-fragment layout (bf16)
    k_pack_w<<<dim3(256, 5), 256, 0, stream>>>(
        (const float*)d_in[3 + 6 * 1], (const float*)d_in[3 + 6 * 2],
        (const float*)d_in[3 + 6 * 3], (const float*)d_in[3 + 6 * 4],
        (const float*)d_in[3 + 6 * 5], wpk);

    for (int l = 0; l < NLAYERS; ++l) {
        const int base = 3 + 6 * l;
        const float* asrc  = (const float*)d_in[base + 1];
        const float* adst  = (const float*)d_in[base + 2];
        const float* bias  = (const float*)d_in[base + 3];
        const float* gamma = (const float*)d_in[base + 4];
        const float* beta  = (const float*)d_in[base + 5];

        if (l == 0) {
            k_gemm0<<<NH_BLOCKS, 256, 0, stream>>>(x_in, (const float*)d_in[3], bufH);
        } else {
            k_cvt_bf16<<<NH_BLOCKS, 256, 0, stream>>>(bufX, bufHbf);
            k_gemm_wmma<<<dim3(N_NODES / 16, 4), 32, 0, stream>>>(
                bufHbf, wpk + (size_t)(l - 1) * 65536, bufH);
        }
        k_dots<<<(N_NODES + 7) / 8, 256, 0, stream>>>(bufH, asrc, adst, aS, aD);
        k_prep<<<NH_BLOCKS, 256, 0, stream>>>(bufX, mOrd, denom);
        k_edge_att<<<NE_BLOCKS, 256, 0, stream>>>(ei, aS, aD, attw, mOrd);
        k_edge_w<<<NE_BLOCKS, 256, 0, stream>>>(ei, attw, mOrd, denom);
        k_aggr<<<(NE_TOT + 7) / 8, 256, 0, stream>>>(ei, attw, denom, bufH, bufX);
        k_bias_relu<<<NH_BLOCKS, 256, 0, stream>>>(bufX, bias, colsum, colsq);
        k_bn_stats<<<128, 256, 0, stream>>>(bufX, colsum, colsq);
        k_bn_apply<<<NH_BLOCKS, 256, 0, stream>>>(bufX, colsum, colsq, gamma, beta);
    }

    k_zero_pool<<<(NG * HD + 255) / 256, 256, 0, stream>>>(pooled, counts);
    k_pool<<<128, 256, 0, stream>>>(bufX, batch, pooled, counts);
    k_pool_div<<<(NG * HD + 255) / 256, 256, 0, stream>>>(pooled, counts);
    k_dense<<<NG, HD, 0, stream>>>(pooled, dW, db, hdbuf);
    k_mu<<<NG, HD, 0, stream>>>(hdbuf, muW, muB, out);
}
